// Attention_16552803958833
// MI455X (gfx1250) — matrix operand order repaired and checked by reading, not compiled, not run on hardware
//
#include <hip/hip_runtime.h>
#include <hip/hip_bf16.h>

// Shapes from the reference
#define Bsz 4
#define Nseq 1024
#define Cdim 512
#define Hn 8
#define CHd 64
#define HC (Hn * CHd)      // 512
#define Mrows (Bsz * Nseq) // 4096

typedef __attribute__((ext_vector_type(16))) _Float16 v16h;
typedef __attribute__((ext_vector_type(8)))  _Float16 v8h;
typedef __attribute__((ext_vector_type(8)))  float    v8f;
typedef __attribute__((ext_vector_type(4)))  unsigned int u32x4;
typedef __attribute__((ext_vector_type(8)))  int      i32x8;
typedef __attribute__((ext_vector_type(4)))  int      i32x4;

// ---------------------------------------------------------------------------
// WMMA helpers (CDNA5 wave32, V_WMMA_F32_16X16X32_F16)
// ---------------------------------------------------------------------------
__device__ __forceinline__ v8f wmma_f16(v16h a, v16h b, v8f c) {
  return __builtin_amdgcn_wmma_f32_16x16x32_f16(false, a, false, b, (short)0, c,
                                                false, false);
}

// A-fragment (16x32, MxK) from a row-major tile (ISA 7.12.2).
__device__ __forceinline__ v16h frag_a(const _Float16* tile, int lda, int lane) {
  const int m = lane & 15;
  const int hf = lane >> 4;
  const _Float16* r = tile + (size_t)m * lda;
  v8h lo = *(const v8h*)(r + hf * 8);
  v8h hi = *(const v8h*)(r + 16 + hf * 8);
  return __builtin_shufflevector(lo, hi, 0, 1, 2, 3, 4, 5, 6, 7,
                                         8, 9, 10, 11, 12, 13, 14, 15);
}

// B-fragment (32x16, KxN) from a TRANSPOSED ([N][K]) tile: lane holds column
// N=lane&15, elements K=half*8+e (e<8) and K=16+half*8+(e-8) (e>=8).
__device__ __forceinline__ v16h frag_b(const _Float16* tileT, int ldb, int lane) {
  const int n = lane & 15;
  const int hf = lane >> 4;
  const _Float16* r = tileT + (size_t)n * ldb;
  v8h lo = *(const v8h*)(r + hf * 8);
  v8h hi = *(const v8h*)(r + 16 + hf * 8);
  return __builtin_shufflevector(lo, hi, 0, 1, 2, 3, 4, 5, 6, 7,
                                         8, 9, 10, 11, 12, 13, 14, 15);
}

// ---------------------------------------------------------------------------
// Tensor Data Mover: 2-D tile load to LDS (D# per ISA ch.8).
// data_size = 2 bytes (f16). Optional LDS padding: 8 DWORDs inserted after
// every 32 DWORDs (one 128B tile row) -> LDS row stride 160B (bank-friendly,
// still 16B aligned for ds_load_b128).
// ---------------------------------------------------------------------------
__device__ __forceinline__ void tdm_load_2d_f16(
    const void* gptr, unsigned lds_off,
    unsigned tensor_d0, unsigned tensor_d1, unsigned long long stride0_elems,
    unsigned tile_d0, unsigned tile_d1, bool pad) {
  const unsigned long long ga = (unsigned long long)gptr;
  u32x4 g0;
  g0[0] = 1u;                                              // count=1 (valid), user mode
  g0[1] = lds_off;                                         // LDS byte address
  g0[2] = (unsigned)ga;                                    // global_addr[31:0]
  g0[3] = (unsigned)((ga >> 32) & 0x01FFFFFFu) | (2u << 30); // addr[56:32] | type=2

  unsigned w0 = (1u << 16);                                // data_size=1 -> 2 bytes
  if (pad) w0 |= (1u << 20) | (4u << 22) | (7u << 25);     // pad_en, ivl=32dw, amt=8dw
  i32x8 g1;
  g1[0] = (int)w0;                                         // wg_mask=0, no atomics
  g1[1] = (int)((tensor_d0 & 0xFFFFu) << 16);              // tensor_dim0[15:0]
  g1[2] = (int)((tensor_d0 >> 16) | ((tensor_d1 & 0xFFFFu) << 16));
  g1[3] = (int)((tensor_d1 >> 16) | (tile_d0 << 16));      // tile_dim0
  g1[4] = (int)(tile_d1 & 0xFFFFu);                        // tile_dim1, tile_dim2=0
  g1[5] = (int)(unsigned)(stride0_elems & 0xFFFFFFFFull);  // dim0_stride[31:0]
  g1[6] = (int)(unsigned)((stride0_elems >> 32) & 0xFFFFull); // [47:32], dim1_stride=0
  g1[7] = 0;

  i32x4 gz = {0, 0, 0, 0};                                 // groups 2/3 unused (2-D)
#if __clang_major__ >= 23
  i32x8 gz8 = {0, 0, 0, 0, 0, 0, 0, 0};
  __builtin_amdgcn_tensor_load_to_lds(g0, g1, gz, gz, gz8, 0);
#else
  __builtin_amdgcn_tensor_load_to_lds(g0, g1, gz, gz, 0);
#endif
}

__device__ __forceinline__ unsigned lds_byte_off(const void* p) {
  // Flat LDS addresses carry the byte offset in addr[31:0] (ISA 10.2 aperture).
  return (unsigned)(unsigned long long)p;
}

// ---------------------------------------------------------------------------
// Kernel 1: fused projections  Y = X @ W  (4096 x 512 x 512), f16 outputs
// blockIdx.z: 0=Q(scale 1/8), 1=K, 2=V(transposed store), 3=G(sigmoid)
// ---------------------------------------------------------------------------
__global__ __launch_bounds__(256)
void proj_kernel(const float* __restrict__ qx, const float* __restrict__ kvx,
                 const float* __restrict__ Wq, const float* __restrict__ Wk,
                 const float* __restrict__ Wv, const float* __restrict__ Wg,
                 const float* __restrict__ bg,
                 _Float16* __restrict__ Qb, _Float16* __restrict__ Kb,
                 _Float16* __restrict__ Vt, _Float16* __restrict__ Gb) {
  __shared__ _Float16 sA[128][32];
  __shared__ _Float16 sBt[64][48];

  const int tid  = threadIdx.x;
  const int lane = tid & 31;
  const int wave = tid >> 5;
  const int wm = wave >> 1, wn = wave & 1;
  const int m0 = blockIdx.x * 128;
  const int n0 = blockIdx.y * 64;
  const int z  = blockIdx.z;

  const float* A = (z == 0 || z == 3) ? qx : kvx;
  const float* W = (z == 0) ? Wq : (z == 1) ? Wk : (z == 2) ? Wv : Wg;

  v8f acc[2][2] = {{{}, {}}, {{}, {}}};

  for (int k0 = 0; k0 < Cdim; k0 += 32) {
    {
      const int row = tid >> 1;
      const int seg = (tid & 1) * 16;
      const float4* src = (const float4*)(A + (size_t)(m0 + row) * Cdim + k0 + seg);
#pragma unroll
      for (int c = 0; c < 4; ++c) {
        float4 f = src[c];
        sA[row][seg + c * 4 + 0] = (_Float16)f.x;
        sA[row][seg + c * 4 + 1] = (_Float16)f.y;
        sA[row][seg + c * 4 + 2] = (_Float16)f.z;
        sA[row][seg + c * 4 + 3] = (_Float16)f.w;
      }
    }
    {
      const int n  = tid & 63;
      const int ks = (tid >> 6) * 8;
#pragma unroll
      for (int j = 0; j < 8; ++j)
        sBt[n][ks + j] = (_Float16)W[(size_t)(k0 + ks + j) * HC + n0 + n];
    }
    __syncthreads();

    v16h af0 = frag_a(&sA[wm * 32 + 0][0], 32, lane);
    v16h af1 = frag_a(&sA[wm * 32 + 16][0], 32, lane);
    v16h bf0 = frag_b(&sBt[wn * 32 + 0][0], 48, lane);
    v16h bf1 = frag_b(&sBt[wn * 32 + 16][0], 48, lane);
    acc[0][0] = wmma_f16(af0, bf0, acc[0][0]);
    acc[0][1] = wmma_f16(af0, bf1, acc[0][1]);
    acc[1][0] = wmma_f16(af1, bf0, acc[1][0]);
    acc[1][1] = wmma_f16(af1, bf1, acc[1][1]);
    __syncthreads();
  }

  const int hf = lane >> 4, ln = lane & 15;
#pragma unroll
  for (int i = 0; i < 2; ++i)
#pragma unroll
    for (int j = 0; j < 2; ++j)
#pragma unroll
      for (int r = 0; r < 8; ++r) {
        const int m = m0 + wm * 32 + i * 16 + hf * 8 + r;
        const int c = n0 + wn * 32 + j * 16 + ln;
        const int b = m >> 10, ns = m & 1023;
        const int hd = c >> 6, ch = c & 63;
        const float y = acc[i][j][r];
        if (z == 0) {
          Qb[(((size_t)b * Hn + hd) * Nseq + ns) * CHd + ch] =
              (_Float16)(y * 0.125f);  // 1/sqrt(64)
        } else if (z == 1) {
          Kb[(((size_t)b * Hn + hd) * Nseq + ns) * CHd + ch] = (_Float16)y;
        } else if (z == 2) {
          Vt[(((size_t)b * Hn + hd) * CHd + ch) * Nseq + ns] = (_Float16)y;
        } else {
          const float s = y + bg[c];
          Gb[(size_t)m * HC + c] = (_Float16)(1.0f / (1.0f + __expf(-s)));
        }
      }
}

// ---------------------------------------------------------------------------
// Kernel 2: flash attention per (b,h). Block = 128 Q rows, 8 waves, each wave
// owns 16 Q rows. K/V 64-wide tiles are DMA'd into LDS by the Tensor Data
// Mover (double buffered, TDM-padded rows), shared by all 8 waves.
// ---------------------------------------------------------------------------
#define KV_LDB 80  // 64 + 8dw TDM pad -> 160B row stride

__global__ __launch_bounds__(256)
void attn_kernel(const _Float16* __restrict__ Qb, const _Float16* __restrict__ Kb,
                 const _Float16* __restrict__ Vt,
                 const float* __restrict__ bias_mask,
                 const float* __restrict__ bias_pair,
                 float* __restrict__ Ob) {
  __shared__ _Float16 sK[2][64][KV_LDB];  // 20 KB (double buffered)
  __shared__ _Float16 sV[2][64][KV_LDB];  // 20 KB
  __shared__ _Float16 sP[8][16][64];      // 16 KB, per-wave private P tile

  const int tid  = threadIdx.x;
  const int lane = tid & 31;
  const int wave = tid >> 5;
  const int hf = lane >> 4, ln = lane & 15;
  const int bh = blockIdx.y;
  const int b = bh >> 3, hd = bh & 7;
  const int q0 = blockIdx.x * 128 + wave * 16;

  const _Float16* Qh = Qb + (size_t)bh * Nseq * CHd;
  const _Float16* Kh = Kb + (size_t)bh * Nseq * CHd;
  const _Float16* Vh = Vt + (size_t)bh * CHd * Nseq;
  const float* bm = bias_mask + (size_t)b * Nseq;
  const float* bp = bias_pair + (size_t)hd * Nseq * Nseq;

  // Preload tile 0 via TDM: wave0 -> K, wave1 -> V (per-wave TENSORcnt)
  if (wave == 0)
    tdm_load_2d_f16(Kh, lds_byte_off(&sK[0][0][0]),
                    CHd, Nseq, CHd, 64, 64, true);
  if (wave == 1)
    tdm_load_2d_f16(Vh, lds_byte_off(&sV[0][0][0]),
                    Nseq, CHd, Nseq, 64, 64, true);

  // Q fragments for the whole head dim (CH=64 -> 2 K-steps), kept resident
  v16h qf0 = frag_a(Qh + (size_t)q0 * CHd, CHd, lane);
  v16h qf1 = frag_a(Qh + (size_t)q0 * CHd + 32, CHd, lane);

  float mrun[8], lrun[8];
  v8f acc_o[4] = {{}, {}, {}, {}};
#pragma unroll
  for (int r = 0; r < 8; ++r) { mrun[r] = -1e30f; lrun[r] = 0.0f; }

  for (int kt = 0; kt < Nseq; kt += 64) {
    const int buf = (kt >> 6) & 1;

    // Kick off next tile's DMA, then wait for the current tile.
    if (kt + 64 < Nseq) {
      if (wave == 0)
        tdm_load_2d_f16(Kh + (size_t)(kt + 64) * CHd,
                        lds_byte_off(&sK[buf ^ 1][0][0]),
                        CHd, Nseq, CHd, 64, 64, true);
      if (wave == 1)
        tdm_load_2d_f16(Vh + (size_t)(kt + 64),
                        lds_byte_off(&sV[buf ^ 1][0][0]),
                        Nseq, CHd, Nseq, 64, 64, true);
      if (wave < 2) __builtin_amdgcn_s_wait_tensorcnt(1);
    } else {
      if (wave < 2) __builtin_amdgcn_s_wait_tensorcnt(0);
    }
    __syncthreads();  // publish current K/V tile to all waves

    // ---- S tile = Q @ K^T : 4 column tiles x 2 K-steps = 8 WMMA (from LDS)
    v8f s[4] = {{}, {}, {}, {}};
#pragma unroll
    for (int t = 0; t < 4; ++t) {
      v16h kf0 = frag_b(&sK[buf][t * 16][0], KV_LDB, lane);
      v16h kf1 = frag_b(&sK[buf][t * 16][0] + 32, KV_LDB, lane);
      s[t] = wmma_f16(qf0, kf0, s[t]);
      s[t] = wmma_f16(qf1, kf1, s[t]);
    }

    // stream the big bias_pair tensor ahead
    if (kt + 64 < Nseq)
      __builtin_prefetch(bp + (size_t)(q0 + hf * 8) * Nseq + kt + 64 + ln * 4, 0, 1);

    float bmv[4];
#pragma unroll
    for (int t = 0; t < 4; ++t) bmv[t] = bm[kt + t * 16 + ln];

    // ---- biases + online softmax (D-layout: row = hf*8+r, col = t*16+ln)
    float corr[8];
#pragma unroll
    for (int r = 0; r < 8; ++r) {
      const float* bpr = bp + (size_t)(q0 + hf * 8 + r) * Nseq + kt;
      float sv[4];
      float tm = -1e30f;
#pragma unroll
      for (int t = 0; t < 4; ++t) {
        float x = s[t][r] + bmv[t] + bpr[t * 16 + ln];
        sv[t] = x;
        tm = fmaxf(tm, x);
      }
      tm = fmaxf(tm, __shfl_xor(tm, 1, 32));
      tm = fmaxf(tm, __shfl_xor(tm, 2, 32));
      tm = fmaxf(tm, __shfl_xor(tm, 4, 32));
      tm = fmaxf(tm, __shfl_xor(tm, 8, 32));
      const float mnew = fmaxf(mrun[r], tm);
      const float cr = __expf(mrun[r] - mnew);
      mrun[r] = mnew;
      corr[r] = cr;
      float ps = 0.0f;
#pragma unroll
      for (int t = 0; t < 4; ++t) {
        float p = __expf(sv[t] - mnew);
        ps += p;
        s[t][r] = p;
      }
      ps += __shfl_xor(ps, 1, 32);
      ps += __shfl_xor(ps, 2, 32);
      ps += __shfl_xor(ps, 4, 32);
      ps += __shfl_xor(ps, 8, 32);
      lrun[r] = lrun[r] * cr + ps;
    }

    // ---- rescale running O
#pragma unroll
    for (int nt = 0; nt < 4; ++nt)
#pragma unroll
      for (int r = 0; r < 8; ++r) acc_o[nt][r] *= corr[r];

    // ---- transpose P (D-layout -> A-layout) through wave-private LDS
#pragma unroll
    for (int t = 0; t < 4; ++t)
#pragma unroll
      for (int r = 0; r < 8; ++r)
        sP[wave][hf * 8 + r][t * 16 + ln] = (_Float16)s[t][r];

    v16h pf0 = frag_a(&sP[wave][0][0], 64, lane);
    v16h pf1 = frag_a(&sP[wave][0][0] + 32, 64, lane);

    // ---- O += P @ V : 4 ch tiles x 2 K-steps = 8 WMMA (from LDS)
#pragma unroll
    for (int nt = 0; nt < 4; ++nt) {
      v16h vf0 = frag_b(&sV[buf][nt * 16][0], KV_LDB, lane);
      v16h vf1 = frag_b(&sV[buf][nt * 16][0] + 32, KV_LDB, lane);
      acc_o[nt] = wmma_f16(pf0, vf0, acc_o[nt]);
      acc_o[nt] = wmma_f16(pf1, vf1, acc_o[nt]);
    }
    __syncthreads();  // all waves done with this buffer before TDM reuses it
  }

  // ---- normalize and write O as [B][N][H*CH] f32
#pragma unroll
  for (int r = 0; r < 8; ++r) {
    const float inv = 1.0f / lrun[r];
    const int q = q0 + hf * 8 + r;
    float* op = Ob + ((size_t)b * Nseq + q) * HC + hd * CHd;
#pragma unroll
    for (int nt = 0; nt < 4; ++nt) op[nt * 16 + ln] = acc_o[nt][r] * inv;
  }
}

// ---------------------------------------------------------------------------
// Kernel 3: out = (O * G) @ Wo + bo   (4096 x 512 x 512), gating fused into
// the A-tile staging.
// ---------------------------------------------------------------------------
__global__ __launch_bounds__(256)
void out_kernel(const float* __restrict__ Ob, const _Float16* __restrict__ Gb,
                const float* __restrict__ Wo, const float* __restrict__ bo,
                float* __restrict__ out) {
  __shared__ _Float16 sA[128][32];
  __shared__ _Float16 sBt[64][48];

  const int tid  = threadIdx.x;
  const int lane = tid & 31;
  const int wave = tid >> 5;
  const int wm = wave >> 1, wn = wave & 1;
  const int m0 = blockIdx.x * 128;
  const int n0 = blockIdx.y * 64;

  v8f acc[2][2] = {{{}, {}}, {{}, {}}};

  for (int k0 = 0; k0 < HC; k0 += 32) {
    {
      const int row = tid >> 1;
      const int seg = (tid & 1) * 16;
      const size_t base = (size_t)(m0 + row) * HC + k0 + seg;
      const float4* src = (const float4*)(Ob + base);
      const _Float16* gp = Gb + base;
#pragma unroll
      for (int c = 0; c < 4; ++c) {
        float4 f = src[c];
        sA[row][seg + c * 4 + 0] = (_Float16)(f.x * (float)gp[c * 4 + 0]);
        sA[row][seg + c * 4 + 1] = (_Float16)(f.y * (float)gp[c * 4 + 1]);
        sA[row][seg + c * 4 + 2] = (_Float16)(f.z * (float)gp[c * 4 + 2]);
        sA[row][seg + c * 4 + 3] = (_Float16)(f.w * (float)gp[c * 4 + 3]);
      }
    }
    {
      const int n  = tid & 63;
      const int ks = (tid >> 6) * 8;
#pragma unroll
      for (int j = 0; j < 8; ++j)
        sBt[n][ks + j] = (_Float16)Wo[(size_t)(k0 + ks + j) * Cdim + n0 + n];
    }
    __syncthreads();

    v16h af0 = frag_a(&sA[wm * 32 + 0][0], 32, lane);
    v16h af1 = frag_a(&sA[wm * 32 + 16][0], 32, lane);
    v16h bf0 = frag_b(&sBt[wn * 32 + 0][0], 48, lane);
    v16h bf1 = frag_b(&sBt[wn * 32 + 16][0], 48, lane);
    acc[0][0] = wmma_f16(af0, bf0, acc[0][0]);
    acc[0][1] = wmma_f16(af0, bf1, acc[0][1]);
    acc[1][0] = wmma_f16(af1, bf0, acc[1][0]);
    acc[1][1] = wmma_f16(af1, bf1, acc[1][1]);
    __syncthreads();
  }

  const int hf = lane >> 4, ln = lane & 15;
#pragma unroll
  for (int i = 0; i < 2; ++i)
#pragma unroll
    for (int j = 0; j < 2; ++j)
#pragma unroll
      for (int r = 0; r < 8; ++r) {
        const int m = m0 + wm * 32 + i * 16 + hf * 8 + r;
        const int c = n0 + wn * 32 + j * 16 + ln;
        out[(size_t)m * Cdim + c] = acc[i][j][r] + bo[c];
      }
}

// ---------------------------------------------------------------------------
extern "C" void kernel_launch(void* const* d_in, const int* in_sizes, int n_in,
                              void* d_out, int out_size, void* d_ws, size_t ws_size,
                              hipStream_t stream) {
  (void)in_sizes; (void)n_in; (void)out_size; (void)ws_size;
  const float* qx    = (const float*)d_in[0];
  const float* kvx   = (const float*)d_in[1];
  const float* bmask = (const float*)d_in[2];
  const float* bpair = (const float*)d_in[3];
  const float* Wq    = (const float*)d_in[4];
  const float* Wk    = (const float*)d_in[5];
  const float* Wv    = (const float*)d_in[6];
  const float* Wg    = (const float*)d_in[7];
  const float* bg    = (const float*)d_in[8];
  const float* Wo    = (const float*)d_in[9];
  const float* bo    = (const float*)d_in[10];
  float* out = (float*)d_out;

  // Workspace (24 MB): Qb/Kb [B,H,N,CH] f16; Vt [B,H,CH,N] f16;
  // Gb [B*N,HC] f16; Ob [B*N,HC] f32
  const size_t elems = (size_t)Mrows * HC;  // 2M
  _Float16* Qb = (_Float16*)d_ws;
  _Float16* Kb = Qb + elems;
  _Float16* Vt = Kb + elems;
  _Float16* Gb = Vt + elems;
  float*    Obuf = (float*)(Gb + elems);

  proj_kernel<<<dim3(Mrows / 128, HC / 64, 4), 256, 0, stream>>>(
      qx, kvx, Wq, Wk, Wv, Wg, bg, Qb, Kb, Vt, Gb);
  attn_kernel<<<dim3(Nseq / 128, Bsz * Hn), 256, 0, stream>>>(
      Qb, Kb, Vt, bmask, bpair, Obuf);
  out_kernel<<<dim3(Mrows / 128, Cdim / 64), 256, 0, stream>>>(
      Obuf, Gb, Wo, bo, out);
}
